// SelfAttention_4329327034632
// MI455X (gfx1250) — matrix operand order repaired and checked by reading, not compiled, not run on hardware
//
#include <hip/hip_runtime.h>
#include <stdint.h>

// B=4, S=4096, C=256, E=256 self-attention, flash-style online softmax.
// All matmuls on v_wmma_f32_16x16x32_bf16 (f32 accumulate).
// Pipeline: (0) x -> bf16 once, (1) QKV GEMM (W tile staged in LDS in
// B-fragment layout, V stored transposed [b][e][s]), (2) flash attention with
// K chunks streamed into LDS by the Tensor Data Mover (double-buffered).

typedef __attribute__((ext_vector_type(16))) __bf16 v16bf;
typedef __attribute__((ext_vector_type(8)))  float  v8f;
typedef __attribute__((ext_vector_type(4)))  unsigned int v4u;
typedef __attribute__((ext_vector_type(8)))  int v8i;
typedef __attribute__((ext_vector_type(4)))  int v4i;

#define BATCH 4
#define SEQ   4096
#define CDIM  256
#define EDIM  256
#define MROWS  (BATCH * SEQ)   // 16384
#define NCHUNK (SEQ / 32)      // 128

// f32 -> bf16 (round-half-up): 2 VALU
__device__ __forceinline__ unsigned short f2bf(float f) {
    union { float f; uint32_t u; } c; c.f = f;
    return (unsigned short)((c.u + 0x8000u) >> 16);
}

// f32x2 -> packed bf16x2 via v_perm_b32: 3 VALU per pair
__device__ __forceinline__ uint32_t pack_bf16(float lo, float hi) {
    union { float f; uint32_t u; } a, b;
    a.f = lo; b.f = hi;
    return __builtin_amdgcn_perm(b.u + 0x8000u, a.u + 0x8000u, 0x07060302u);
}

union FragBF {
    v16bf v;
    uint4 q2[2];
    uint32_t d[8];
    unsigned short u[16];
};

// TDM: load a 2D tile (t0 x t1 elems of 2 bytes, row stride stride0 elems)
// from global into LDS at lds_addr. D# layout per CDNA5 ISA ch.8.
__device__ __forceinline__ void tdm_load_2d(unsigned lds_addr, const void* gptr,
                                            unsigned d0, unsigned d1,
                                            unsigned t0, unsigned t1,
                                            unsigned long long stride0)
{
    const unsigned long long ga = (unsigned long long)(size_t)gptr;
    v4u g0 = { 1u,                                    // count=1, user mode
               lds_addr,                              // lds_addr [63:32]
               (unsigned)ga,                          // global_addr [95:64]
               (unsigned)((ga >> 32) & 0x01FFFFFFu)   // global_addr [120:96]
                   | 0x80000000u };                   // type=2 [127:126]
    v8i g1;
    g1[0] = 0x10000;                                  // data_size=1 (2 bytes)
    g1[1] = (int)((d0 & 0xFFFFu) << 16);              // tensor_dim0 [79:48]
    g1[2] = (int)((d0 >> 16) | ((d1 & 0xFFFFu) << 16)); // tensor_dim1 [111:80]
    g1[3] = (int)((d1 >> 16) | (t0 << 16));           // tile_dim0 [127:112]
    g1[4] = (int)t1;                                  // tile_dim1 [143:128]
    g1[5] = (int)(unsigned)stride0;                   // dim0_stride [207:160]
    g1[6] = (int)(unsigned)(stride0 >> 32);
    g1[7] = 0;
    v4i gz = { 0, 0, 0, 0 };
#if defined(__clang_major__) && (__clang_major__ >= 23)
    v8i gz8 = { 0, 0, 0, 0, 0, 0, 0, 0 };
    __builtin_amdgcn_tensor_load_to_lds(g0, g1, gz, gz, gz8, 0);
#else
    __builtin_amdgcn_tensor_load_to_lds(g0, g1, gz, gz, 0);
#endif
}

// ---------------------------------------------------------------------------
// Kernel 0: one-shot x -> bf16 conversion (8 elements/thread).
// ---------------------------------------------------------------------------
__global__ __launch_bounds__(256) void cvt_kernel(
    const float* __restrict__ x, unsigned short* __restrict__ xbf)
{
    const size_t i = ((size_t)blockIdx.x * 256 + threadIdx.x) * 8;
    float4 f0 = *(const float4*)(x + i);
    float4 f1 = *(const float4*)(x + i + 4);
    uint4 o;
    o.x = pack_bf16(f0.x, f0.y);
    o.y = pack_bf16(f0.z, f0.w);
    o.z = pack_bf16(f1.x, f1.y);
    o.w = pack_bf16(f1.z, f1.w);
    *(uint4*)(xbf + i) = o;
}

// ---------------------------------------------------------------------------
// Kernel 1: QKV projection. One block owns one (which,nTile) 256x16 W tile
// (staged once in LDS, pre-swizzled to B-fragment layout) and 8 m-tiles
// (one per wave). Q,K stored row-major bf16; V stored transposed [b][e][s].
// ---------------------------------------------------------------------------
__global__ __launch_bounds__(256, 1) void qkv_kernel(
    const unsigned short* __restrict__ xbf,
    const float* __restrict__ Wq, const float* __restrict__ bq,
    const float* __restrict__ Wk, const float* __restrict__ bk,
    const float* __restrict__ Wv, const float* __restrict__ bv,
    unsigned short* __restrict__ Qws,
    unsigned short* __restrict__ Kws,
    unsigned short* __restrict__ Vt)
{
    // [8 kc][2 half][16 n][8 dwords] = 2048 dwords = 8 KB, B-frag layout
    __shared__ __align__(16) uint32_t Wl[2048];

    const int tid    = threadIdx.x;
    const int lane   = tid & 31;
    const int wave   = tid >> 5;
    const int ng     = blockIdx.x >> 7;          // 0..47
    const int mGroup = blockIdx.x & 127;         // 0..127
    const int which  = ng >> 4;                  // 0=Q, 1=K, 2=V
    const int nTile  = ng & 15;
    const int mTile  = mGroup * 8 + wave;        // 0..1023

    const float* W; const float* bias;
    if (which == 0)      { W = Wq; bias = bq; }
    else if (which == 1) { W = Wk; bias = bk; }
    else                 { W = Wv; bias = bv; }

    // ---- cooperative W-tile stage: thread -> (kcF, hF, nF) slot ----
    {
        const int kcF = tid >> 5, hF = (tid >> 4) & 1, nF = tid & 15;
        const int r0   = kcF * 32 + hF * 16;
        const int colF = nTile * 16 + nF;
        #pragma unroll
        for (int i = 0; i < 8; ++i) {
            const float lo = W[(size_t)(r0 + 2 * i)     * CDIM + colF];
            const float hi = W[(size_t)(r0 + 2 * i + 1) * CDIM + colF];
            Wl[tid * 8 + i] = pack_bf16(lo, hi);
        }
    }
    __syncthreads();

    const int m     = lane & 15;
    const int half  = lane >> 4;
    const int kbase = half * 16;                 // A-matrix K contiguous
    const int row   = mTile * 16 + m;            // A row (this lane)
    const int col   = nTile * 16 + m;            // B/N column (this lane)

    v8f acc = {};
    #pragma unroll
    for (int kc = 0; kc < 8; ++kc) {
        FragBF a, b;
        const unsigned short* xp = xbf + (size_t)row * CDIM + kc * 32 + kbase;
        a.q2[0] = *(const uint4*)(xp);
        a.q2[1] = *(const uint4*)(xp + 8);
        const uint4* wp = (const uint4*)(Wl + ((kc * 2 + half) * 16 + m) * 8);
        b.q2[0] = wp[0];
        b.q2[1] = wp[1];
        acc = __builtin_amdgcn_wmma_f32_16x16x32_bf16(
                  false, a.v, false, b.v, (short)0, acc, false, false);
    }

    // ---- epilogue: bias + store bf16 ----
    const float bcol = bias[col];
    if (which == 2) {
        const int rowb = mTile * 16;
        const int bb   = rowb >> 12;             // / SEQ
        #pragma unroll
        for (int r = 0; r < 8; ++r) {
            const int M = r + half * 8;
            const int s = (rowb + M) & (SEQ - 1);
            Vt[((size_t)bb * EDIM + col) * SEQ + s] = f2bf(acc[r] + bcol);
        }
    } else {
        unsigned short* outp = (which == 0) ? Qws : Kws;
        #pragma unroll
        for (int r = 0; r < 8; ++r) {
            const int M = r + half * 8;
            outp[(size_t)(mTile * 16 + M) * CDIM + col] = f2bf(acc[r] + bcol);
        }
    }
}

// ---------------------------------------------------------------------------
// Kernel 2: flash attention. 8 waves/block share one batch; K chunks are
// streamed into LDS by the TDM (double-buffered), scores = Q*K^T (WMMA),
// online softmax, O += P*V (WMMA, V from transposed Vt), sigmoid epilogue.
// ---------------------------------------------------------------------------
__global__ __launch_bounds__(256, 1) void attn_kernel(
    const unsigned short* __restrict__ Qws,
    const unsigned short* __restrict__ Kws,
    const unsigned short* __restrict__ Vt,
    float* __restrict__ out)
{
    __shared__ __align__(16) unsigned short Kc[2][32 * CDIM];  // 2 x 16 KB
    __shared__ __align__(16) unsigned short Pbuf[8][16 * 32];  // 8 KB

    const int lane  = threadIdx.x & 31;
    const int wave  = threadIdx.x >> 5;
    const int w     = blockIdx.x * 8 + wave;     // 1024 waves total
    const int b     = w >> 8;                    // uniform per block
    const int qt    = w & 255;
    const size_t bOff = (size_t)b * SEQ;
    const int m     = lane & 15;
    const int half  = lane >> 4;
    const int kbase = half * 16;
    const int q0    = qt * 16;

    const unsigned short* kgl   = Kws + bOff * CDIM;       // this batch's K
    const unsigned short* vbase = Vt + (size_t)b * EDIM * SEQ;

    // Prologue: TDM chunks 0 and 1 into the two K buffers.
    if (wave == 0) {
        tdm_load_2d((unsigned)(size_t)&Kc[0][0], kgl,
                    CDIM, 32, CDIM, 32, CDIM);
        tdm_load_2d((unsigned)(size_t)&Kc[1][0], kgl + 32 * CDIM,
                    CDIM, 32, CDIM, 32, CDIM);
    }

    // Preload Q as 8 A-fragments (K = 256 in 32-chunks).
    FragBF qf[8];
    #pragma unroll
    for (int kc = 0; kc < 8; ++kc) {
        const unsigned short* qp = Qws + (bOff + q0 + m) * CDIM + kc * 32 + kbase;
        qf[kc].q2[0] = *(const uint4*)(qp);
        qf[kc].q2[1] = *(const uint4*)(qp + 8);
    }

    const v8f vzero = {};
    v8f acc[16];
    #pragma unroll
    for (int e = 0; e < 16; ++e) acc[e] = vzero;
    float mrow[8], lrow[8];
    #pragma unroll
    for (int r = 0; r < 8; ++r) { mrow[r] = -1e30f; lrow[r] = 0.0f; }

    unsigned short* P = Pbuf[wave];

    for (int ci = 0; ci < NCHUNK; ++ci) {
        const int bi = ci & 1;
        const int j  = ci * 32;

        // Wait for chunk ci's TDM (in-order; chunk ci+1 may stay in flight).
        if (wave == 0) {
            if (ci + 1 < NCHUNK) __builtin_amdgcn_s_wait_tensorcnt(1);
            else                 __builtin_amdgcn_s_wait_tensorcnt(0);
        }
        __syncthreads();
        const unsigned short* Kcur = &Kc[bi][0];

        // ---- scores: two 16-key subtiles, K-dim = 256, K tile from LDS ----
        v8f s[2];
        #pragma unroll
        for (int st = 0; st < 2; ++st) {
            v8f sa = vzero;
            const int krl = st * 16 + m;         // key row within chunk
            #pragma unroll
            for (int kc = 0; kc < 8; ++kc) {
                FragBF kb;
                const unsigned short* kp = Kcur + krl * CDIM + kc * 32 + half * 16;
                kb.q2[0] = *(const uint4*)(kp);
                kb.q2[1] = *(const uint4*)(kp + 8);
                sa = __builtin_amdgcn_wmma_f32_16x16x32_bf16(
                         false, qf[kc].v, false, kb.v, (short)0, sa, false, false);
            }
            s[st] = sa;
        }

        // ---- online softmax (per C/D row r; 16-lane butterfly within half) --
        float scale[8];
        #pragma unroll
        for (int r = 0; r < 8; ++r) {
            float cmax = fmaxf(s[0][r], s[1][r]);
            #pragma unroll
            for (int d = 1; d < 16; d <<= 1)
                cmax = fmaxf(cmax, __shfl_xor(cmax, d, 32));
            const float mnew = fmaxf(mrow[r], cmax);
            const float p0 = __expf(s[0][r] - mnew);
            const float p1 = __expf(s[1][r] - mnew);
            float rsum = p0 + p1;
            #pragma unroll
            for (int d = 1; d < 16; d <<= 1)
                rsum += __shfl_xor(rsum, d, 32);
            const float sc = __expf(mrow[r] - mnew);
            scale[r] = sc;
            lrow[r]  = lrow[r] * sc + rsum;
            mrow[r]  = mnew;
            const int mr = r + half * 8;
            P[mr * 32 + m]      = f2bf(p0);
            P[mr * 32 + 16 + m] = f2bf(p1);
        }

        // rescale O accumulators
        #pragma unroll
        for (int e = 0; e < 16; ++e) {
            #pragma unroll
            for (int r = 0; r < 8; ++r)
                acc[e][r] = acc[e][r] * scale[r];
        }

        // wave-local LDS store->load ordering (DS is in-order per wave)
        asm volatile("s_wait_dscnt 0" ::: "memory");

        FragBF pf;
        const unsigned short* pp = P + m * 32 + kbase;
        pf.q2[0] = *(const uint4*)(pp);
        pf.q2[1] = *(const uint4*)(pp + 8);

        // ---- O += P @ V : B-fragments are contiguous b128 loads from Vt ----
        #pragma unroll
        for (int e = 0; e < 16; ++e) {
            FragBF vb;
            const unsigned short* vp =
                vbase + (size_t)(e * 16 + m) * SEQ + j + half * 16;
            vb.q2[0] = *(const uint4*)(vp);
            vb.q2[1] = *(const uint4*)(vp + 8);
            acc[e] = __builtin_amdgcn_wmma_f32_16x16x32_bf16(
                         false, pf.v, false, vb.v, (short)0, acc[e], false, false);
        }

        // All waves done reading Kc[bi]; refill it with chunk ci+2.
        __syncthreads();
        if (wave == 0 && ci + 2 < NCHUNK) {
            tdm_load_2d((unsigned)(size_t)&Kc[bi][0],
                        kgl + (size_t)(ci + 2) * 32 * CDIM,
                        CDIM, 32, CDIM, 32, CDIM);
        }
    }

    // ---- epilogue: normalize, sigmoid, store fp32 ----
    #pragma unroll
    for (int r = 0; r < 8; ++r) {
        const float inv = 1.0f / lrow[r];
        const int M = r + half * 8;
        #pragma unroll
        for (int e = 0; e < 16; ++e) {
            const float v  = acc[e][r] * inv;
            const float sg = 1.0f / (1.0f + __expf(-v));
            out[(bOff + q0 + M) * EDIM + e * 16 + m] = sg;
        }
    }
}

// ---------------------------------------------------------------------------
extern "C" void kernel_launch(void* const* d_in, const int* in_sizes, int n_in,
                              void* d_out, int out_size, void* d_ws, size_t ws_size,
                              hipStream_t stream) {
    const float* x  = (const float*)d_in[0];
    const float* Wq = (const float*)d_in[1];
    const float* bq = (const float*)d_in[2];
    const float* Wk = (const float*)d_in[3];
    const float* bk = (const float*)d_in[4];
    const float* Wv = (const float*)d_in[5];
    const float* bv = (const float*)d_in[6];
    float* out = (float*)d_out;

    const size_t MC = (size_t)MROWS * CDIM;      // 16384*256 elements
    unsigned short* Qws = (unsigned short*)d_ws;
    unsigned short* Kws = Qws + MC;
    unsigned short* Vt  = Kws + MC;              // transposed V: [B][E][S]
    unsigned short* xbf = Vt + MC;               // bf16 copy of x (32 MB total)

    cvt_kernel<<<(int)(MC / (256 * 8)), 256, 0, stream>>>(x, xbf);
    qkv_kernel<<<6144, 256, 0, stream>>>(xbf, Wq, bq, Wk, bk, Wv, bv, Qws, Kws, Vt);
    // 4 batches * 256 q-tiles = 1024 waves; 8 waves/block -> 128 blocks
    attn_kernel<<<128, 256, 0, stream>>>(Qws, Kws, Vt, out);
}
